// AttentionMechanism_51049981281163
// MI455X (gfx1250) — matrix operand order repaired and checked by reading, not compiled, not run on hardware
//
#include <hip/hip_runtime.h>

#define B_    2
#define S_    2048
#define D_    1024
#define H_    16
#define HD_   64
#define BH_   (B_ * H_)
#define ROWS_ (B_ * S_)
#define SCALE_ 0.125f

typedef __attribute__((ext_vector_type(16))) __bf16 v16bf;
typedef __attribute__((ext_vector_type(8)))  __bf16 v8bf;
typedef __attribute__((ext_vector_type(8)))  float  v8f;

union Frag { v16bf v; v8bf h[2]; };

__device__ __forceinline__ unsigned short f2bfu(float x) {
    unsigned u = __builtin_bit_cast(unsigned, x);
    unsigned r = (u + 0x7FFFu + ((u >> 16) & 1u)) >> 16;
    return (unsigned short)r;
}

__device__ __forceinline__ v8bf ld8(const unsigned short* p) {
    return *reinterpret_cast<const v8bf*>(p);
}

// A fragment (16x32 bf16): row = lane&15; K elements split per half-wave:
// elems 0..7 -> K = hi*8 + 0..7 ; elems 8..15 -> K = 16 + hi*8 + 0..7
__device__ __forceinline__ v16bf load_a(const unsigned short* rowptr, int hi) {
    Frag f;
    f.h[0] = ld8(rowptr + hi * 8);
    f.h[1] = ld8(rowptr + 16 + hi * 8);
    return f.v;
}

// B fragment (32x16 bf16): col = lane&15; elems 0..7 -> K = hi*8 + 0..7 ; elems 8..15 -> K = 16 + hi*8 + 0..7
__device__ __forceinline__ v16bf load_b(const unsigned short* colptr, int hi) {
    Frag f;
    f.h[0] = ld8(colptr + hi * 8);
    f.h[1] = ld8(colptr + 16 + hi * 8);
    return f.v;
}

__device__ __forceinline__ v8f wmma_bf16(v16bf a, v16bf b, v8f c) {
    return __builtin_amdgcn_wmma_f32_16x16x32_bf16(
        false, a, false, b, (short)0, c, false, false);
}

// ---------------- conversion kernels ----------------

__global__ void cvt_f32_bf16(const float* __restrict__ in,
                             unsigned short* __restrict__ out, int n) {
    int i = blockIdx.x * blockDim.x + threadIdx.x;
    if (i < n) out[i] = f2bfu(in[i]);
}

// out[n*D + k] = bf16(in[k*D + n])   (weights stored transposed, N x K)
__global__ void transpose_w(const float* __restrict__ in,
                            unsigned short* __restrict__ out) {
    int i = blockIdx.x * blockDim.x + threadIdx.x;
    int n = i >> 10;
    int k = i & (D_ - 1);
    out[i] = f2bfu(in[k * D_ + n]);
}

// ---------------- fused QKV projection ----------------
// 2x2 register-blocked: each wave computes a 32x32 C block (4 WMMA tiles).
// grid (ROWS_/32, D_/128, 3), block 128 (4 waves side by side along N)
__global__ __launch_bounds__(128)
void qkv_kernel(const unsigned short* __restrict__ xb,
                const unsigned short* __restrict__ Wqt,
                const unsigned short* __restrict__ Wkt,
                const unsigned short* __restrict__ Wvt,
                const float* __restrict__ bq,
                const float* __restrict__ bk,
                const float* __restrict__ bv,
                unsigned short* __restrict__ Qb,
                unsigned short* __restrict__ Kb,
                unsigned short* __restrict__ Vt) {
    const int lane = threadIdx.x & 31;
    const int w    = threadIdx.x >> 5;
    const int l    = lane & 15;
    const int hi   = lane >> 4;
    const int z    = blockIdx.z;

    const unsigned short* Wt   = (z == 0) ? Wqt : (z == 1) ? Wkt : Wvt;
    const float*          bias = (z == 0) ? bq : (z == 1) ? bk : bv;

    const int gm0 = blockIdx.x << 5;                  // 32 rows
    const int gn0 = (blockIdx.y * 4 + w) << 5;        // 32 cols

    const unsigned short* arow0 = xb + (size_t)(gm0 + l) * D_;
    const unsigned short* arow1 = xb + (size_t)(gm0 + 16 + l) * D_;
    const unsigned short* bcol0 = Wt + (size_t)(gn0 + l) * D_;
    const unsigned short* bcol1 = Wt + (size_t)(gn0 + 16 + l) * D_;

    v8f cc[2][2] = {};
    for (int kk = 0; kk < D_; kk += 32) {
        __builtin_prefetch(arow0 + kk + 512, 0, 1);
        __builtin_prefetch(bcol0 + kk + 512, 0, 1);
        v16bf a0 = load_a(arow0 + kk, hi);
        v16bf a1 = load_a(arow1 + kk, hi);
        v16bf b0 = load_b(bcol0 + kk, hi);
        v16bf b1 = load_b(bcol1 + kk, hi);
        cc[0][0] = wmma_bf16(a0, b0, cc[0][0]);
        cc[0][1] = wmma_bf16(a0, b1, cc[0][1]);
        cc[1][0] = wmma_bf16(a1, b0, cc[1][0]);
        cc[1][1] = wmma_bf16(a1, b1, cc[1][1]);
    }

#pragma unroll
    for (int mi = 0; mi < 2; mi++) {
#pragma unroll
        for (int ni = 0; ni < 2; ni++) {
            const int gn   = gn0 + ni * 16 + l;
            const float bv_ = bias[gn];
            const int h  = gn >> 6;
            const int hd = gn & (HD_ - 1);
#pragma unroll
            for (int r = 0; r < 8; r++) {
                const int gm = gm0 + mi * 16 + r + hi * 8;
                const int b_ = gm >> 11;          // / S_
                const int s  = gm & (S_ - 1);
                const unsigned short val = f2bfu(cc[mi][ni][r] + bv_);
                if (z == 2) {
                    // V transposed: (b,h,hd,s)
                    Vt[((size_t)((b_ * H_ + h) * HD_ + hd) << 11) + s] = val;
                } else {
                    unsigned short* dst = (z == 0) ? Qb : Kb;
                    dst[((size_t)((b_ * H_ + h) * S_ + s) << 6) + hd] = val;
                }
            }
        }
    }
}

// ---------------- attention (flash-style, 16 queries per wave) ----------------
// Pass 1: row max only (no exp).  Pass 2: exp, row-sum, and ctx WMMA; sums
// reduced by a 16-lane butterfly afterwards (exact softmax).
// grid (BH_*(S_/16)/4), block 128
__global__ __launch_bounds__(128)
void attn_kernel(const unsigned short* __restrict__ Qb,
                 const unsigned short* __restrict__ Kb,
                 const unsigned short* __restrict__ Vt,
                 const int* __restrict__ mask,
                 unsigned short* __restrict__ Ctx) {
    __shared__ __align__(16) unsigned short Pl[4][16 * 32];

    const int lane = threadIdx.x & 31;
    const int w    = threadIdx.x >> 5;
    const int l    = lane & 15;
    const int hi   = lane >> 4;
    const int task = blockIdx.x * 4 + w;
    const int qt   = task & 127;
    const int bh   = task >> 7;
    const int b    = bh >> 4;
    const int h    = bh & 15;
    const int q0   = qt << 4;

    const unsigned short* qrow = Qb + ((size_t)(bh * S_ + q0 + l) << 6);
    const v16bf qa0 = load_a(qrow, hi);
    const v16bf qa1 = load_a(qrow + 32, hi);
    const int* mrow = mask + b * S_;

    // ---- pass 1: per-lane running max, then butterfly max across 16 lanes ----
    float pm[8];
#pragma unroll
    for (int r = 0; r < 8; r++) pm[r] = -1e30f;

    for (int j = 0; j < S_ / 16; j++) {
        const int key = (j << 4) + l;
        const unsigned short* kcol = Kb + ((size_t)(bh * S_ + key) << 6);
        v8f s = {};
        s = wmma_bf16(qa0, load_b(kcol, hi), s);
        s = wmma_bf16(qa1, load_b(kcol + 32, hi), s);
        const bool valid = (mrow[key] != 0);
#pragma unroll
        for (int r = 0; r < 8; r++)
            pm[r] = fmaxf(pm[r], valid ? s[r] * SCALE_ : -1e30f);
    }
#pragma unroll
    for (int off = 1; off < 16; off <<= 1) {
#pragma unroll
        for (int r = 0; r < 8; r++)
            pm[r] = fmaxf(pm[r], __shfl_xor(pm[r], off, 16));
    }

    // ---- pass 2: P = exp(S - max); accumulate row-sum and ctx += P @ V ----
    float ls[8];
#pragma unroll
    for (int r = 0; r < 8; r++) ls[r] = 0.f;
    v8f oacc[4] = {};
    unsigned short* pbuf = Pl[w];

    for (int jj = 0; jj < S_ / 32; jj++) {
#pragma unroll
        for (int t = 0; t < 2; t++) {
            const int key = (jj << 5) + (t << 4) + l;
            const unsigned short* kcol = Kb + ((size_t)(bh * S_ + key) << 6);
            v8f s = {};
            s = wmma_bf16(qa0, load_b(kcol, hi), s);
            s = wmma_bf16(qa1, load_b(kcol + 32, hi), s);
            const bool valid = (mrow[key] != 0);
#pragma unroll
            for (int r = 0; r < 8; r++) {
                float v = valid ? s[r] * SCALE_ : -1e30f;
                float p = __expf(v - pm[r]);
                ls[r] += p;
                pbuf[(r + hi * 8) * 32 + (t << 4) + l] = f2bfu(p);
            }
        }
        asm volatile("s_wait_dscnt 0" ::: "memory");
        const v16bf pa = load_a(pbuf + l * 32, hi);
        const int key0 = jj << 5;
#pragma unroll
        for (int t = 0; t < 4; t++) {
            const unsigned short* vcol =
                Vt + ((size_t)(bh * HD_ + (t << 4) + l) << 11) + key0;
            oacc[t] = wmma_bf16(pa, load_b(vcol, hi), oacc[t]);
        }
    }

    // butterfly-sum the per-lane partial row sums, normalize, store
#pragma unroll
    for (int off = 1; off < 16; off <<= 1) {
#pragma unroll
        for (int r = 0; r < 8; r++)
            ls[r] += __shfl_xor(ls[r], off, 16);
    }
    float rc[8];
#pragma unroll
    for (int r = 0; r < 8; r++) rc[r] = 1.0f / ls[r];

#pragma unroll
    for (int r = 0; r < 8; r++) {
        const int srow = q0 + r + hi * 8;
        unsigned short* dst = Ctx + ((size_t)(b * S_ + srow) << 10) + h * HD_;
#pragma unroll
        for (int t = 0; t < 4; t++)
            dst[(t << 4) + l] = f2bfu(oacc[t][r] * rc[r]);
    }
}

// ---------------- output projection ----------------
// 2x2 register-blocked, f32 output.  grid (ROWS_/32, D_/128), block 128
__global__ __launch_bounds__(128)
void out_gemm(const unsigned short* __restrict__ ctxb,
              const unsigned short* __restrict__ Wot,
              const float* __restrict__ bo,
              float* __restrict__ out) {
    const int lane = threadIdx.x & 31;
    const int w    = threadIdx.x >> 5;
    const int l    = lane & 15;
    const int hi   = lane >> 4;
    const int gm0  = blockIdx.x << 5;
    const int gn0  = (blockIdx.y * 4 + w) << 5;

    const unsigned short* arow0 = ctxb + (size_t)(gm0 + l) * D_;
    const unsigned short* arow1 = ctxb + (size_t)(gm0 + 16 + l) * D_;
    const unsigned short* bcol0 = Wot + (size_t)(gn0 + l) * D_;
    const unsigned short* bcol1 = Wot + (size_t)(gn0 + 16 + l) * D_;

    v8f cc[2][2] = {};
    for (int kk = 0; kk < D_; kk += 32) {
        __builtin_prefetch(arow0 + kk + 512, 0, 1);
        __builtin_prefetch(bcol0 + kk + 512, 0, 1);
        v16bf a0 = load_a(arow0 + kk, hi);
        v16bf a1 = load_a(arow1 + kk, hi);
        v16bf b0 = load_b(bcol0 + kk, hi);
        v16bf b1 = load_b(bcol1 + kk, hi);
        cc[0][0] = wmma_bf16(a0, b0, cc[0][0]);
        cc[0][1] = wmma_bf16(a0, b1, cc[0][1]);
        cc[1][0] = wmma_bf16(a1, b0, cc[1][0]);
        cc[1][1] = wmma_bf16(a1, b1, cc[1][1]);
    }

#pragma unroll
    for (int mi = 0; mi < 2; mi++) {
#pragma unroll
        for (int ni = 0; ni < 2; ni++) {
            const int gn    = gn0 + ni * 16 + l;
            const float bv_ = bo[gn];
#pragma unroll
            for (int r = 0; r < 8; r++) {
                const int gm = gm0 + mi * 16 + r + hi * 8;
                out[(size_t)gm * D_ + gn] = cc[mi][ni][r] + bv_;
            }
        }
    }
}

// ---------------- host launcher ----------------

extern "C" void kernel_launch(void* const* d_in, const int* in_sizes, int n_in,
                              void* d_out, int out_size, void* d_ws, size_t ws_size,
                              hipStream_t stream) {
    (void)in_sizes; (void)n_in; (void)out_size; (void)ws_size;

    const float* x   = (const float*)d_in[0];
    const int*  mask = (const int*)d_in[1];
    const float* Wq  = (const float*)d_in[2];
    const float* bq  = (const float*)d_in[3];
    const float* Wk  = (const float*)d_in[4];
    const float* bk  = (const float*)d_in[5];
    const float* Wv  = (const float*)d_in[6];
    const float* bv  = (const float*)d_in[7];
    const float* Wo  = (const float*)d_in[8];
    const float* bo  = (const float*)d_in[9];
    float* out = (float*)d_out;

    unsigned short* ws = (unsigned short*)d_ws;
    const size_t E  = (size_t)ROWS_ * D_;   // activation-sized bf16 buffer
    const size_t WE = (size_t)D_ * D_;      // weight-sized bf16 buffer

    unsigned short* xb  = ws;
    unsigned short* Wqt = xb  + E;
    unsigned short* Wkt = Wqt + WE;
    unsigned short* Wvt = Wkt + WE;
    unsigned short* Wot = Wvt + WE;
    unsigned short* Qb  = Wot + WE;
    unsigned short* Kb  = Qb  + E;
    unsigned short* Vt  = Kb  + E;
    unsigned short* Ctx = Vt  + E;

    cvt_f32_bf16<<<(int)(E / 256), 256, 0, stream>>>(x, xb, (int)E);
    transpose_w<<<(int)(WE / 256), 256, 0, stream>>>(Wq, Wqt);
    transpose_w<<<(int)(WE / 256), 256, 0, stream>>>(Wk, Wkt);
    transpose_w<<<(int)(WE / 256), 256, 0, stream>>>(Wv, Wvt);
    transpose_w<<<(int)(WE / 256), 256, 0, stream>>>(Wo, Wot);

    qkv_kernel<<<dim3(ROWS_ / 32, D_ / 128, 3), 128, 0, stream>>>(
        xb, Wqt, Wkt, Wvt, bq, bk, bv, Qb, Kb, Vt);

    attn_kernel<<<dim3(BH_ * (S_ / 16) / 4), 128, 0, stream>>>(
        Qb, Kb, Vt, mask, Ctx);

    out_gemm<<<dim3(ROWS_ / 32, D_ / 128), 128, 0, stream>>>(Ctx, Wot, bo, out);
}
